// GCN_9887014715840
// MI455X (gfx1250) — matrix operand order repaired and checked by reading, not compiled, not run on hardware
//
#include <hip/hip_runtime.h>
#include <hip/hip_bf16.h>

#define NNODES 100000
#define BN_EPS 1e-5f

typedef __attribute__((ext_vector_type(16))) __bf16 v16bf;
typedef __attribute__((ext_vector_type(8)))  __bf16 v8bf;
typedef __attribute__((ext_vector_type(8)))  float  v8f;

static __device__ __forceinline__ __bf16 f2bf(float f) {
    unsigned u = __builtin_bit_cast(unsigned, f);
    unsigned r = (u + 0x7FFFu + ((u >> 16) & 1u)) >> 16;   // round-to-nearest-even
    unsigned short s = (unsigned short)r;
    return __builtin_bit_cast(__bf16, s);
}

// ---------------------------------------------------------------------------
// fp32 -> bf16 bulk convert
// ---------------------------------------------------------------------------
__global__ void cvt_f32_bf16(const float* __restrict__ src, __bf16* __restrict__ dst,
                             long n) {
    long i = (long)blockIdx.x * blockDim.x + threadIdx.x;
    long stride = (long)gridDim.x * blockDim.x;
    for (; i < n; i += stride) dst[i] = f2bf(src[i]);
}

// ---------------------------------------------------------------------------
// Pack W [K,N] fp32 row-major -> bf16 WMMA B-fragment layout:
//   Wpk[((kt*tilesN + nt)*32 + lane)*16 + i] = W[kt*32 + i + (lane>>4)*16][nt*16 + (lane&15)]
// so the GEMM reads one contiguous v16bf per lane per k-step.
// ---------------------------------------------------------------------------
__global__ void pack_w_bf16(const float* __restrict__ W, __bf16* __restrict__ Wpk,
                            int K, int N) {
    int total = (K >> 5) * (N >> 4) * 32 * 16;
    int idx = blockIdx.x * blockDim.x + threadIdx.x;
    if (idx >= total) return;
    int i    = idx & 15;
    int lane = (idx >> 4) & 31;
    int t    = idx >> 9;
    int tilesN = N >> 4;
    int nt = t % tilesN;
    int kt = t / tilesN;
    int k  = kt * 32 + i + ((lane >> 4) << 4);
    int n  = nt * 16 + (lane & 15);
    Wpk[idx] = f2bf(W[(long)k * N + n]);
}

// ---------------------------------------------------------------------------
// WMMA GEMM: C[M,N] = A[M,K] (bf16 row-major) x Bpk (packed bf16), fp32 accum.
// One 16x16 C tile per wave, 8 waves per block. M,N,K multiples of 16/16/32.
// ---------------------------------------------------------------------------
__global__ void wmma_gemm_bf16(const __bf16* __restrict__ A,
                               const __bf16* __restrict__ Bpk,
                               float* __restrict__ C,
                               int M, int N, int K) {
    const int lane   = threadIdx.x & 31;
    const int wave   = threadIdx.x >> 5;
    const int tilesN = N >> 4;
    const int numTiles = (M >> 4) * tilesN;
    const int tileId = blockIdx.x * 8 + wave;
    if (tileId >= numTiles) return;
    const int tm = tileId / tilesN;
    const int tn = tileId % tilesN;

    const int half = lane >> 4;        // lane half selects K sub-block
    const int mrow = lane & 15;        // A-matrix row within tile
    const __bf16* arow = A + (long)(tm * 16 + mrow) * K;

    v8f acc = {};
    const int kTiles = K >> 5;
    for (int kt = 0; kt < kTiles; ++kt) {
        const __bf16* aptr = arow + kt * 32;
        __builtin_prefetch(aptr + 32, 0, 1);           // next K-step of this row
        // A layout matches packed B: lanes0-15 hold K{0..15}, lanes16-31 K{16..31}
        v8bf a0 = *(const v8bf*)(aptr + half * 16);
        v8bf a1 = *(const v8bf*)(aptr + half * 16 + 8);
        v16bf afrag = __builtin_shufflevector(a0, a1, 0, 1, 2, 3, 4, 5, 6, 7,
                                                      8, 9, 10, 11, 12, 13, 14, 15);
        v16bf bfrag = *(const v16bf*)(Bpk + ((long)(kt * tilesN + tn) * 32 + lane) * 16);
        acc = __builtin_amdgcn_wmma_f32_16x16x32_bf16(
            /*neg_a=*/false, afrag, /*neg_b=*/false, bfrag,
            /*c_mod=*/(short)0, acc, /*reuse_a=*/false, /*reuse_b=*/false);
    }
    // C/D layout: VGPR r -> M = r + half*8, N = lane&15
    float* cptr = C + (long)(tm * 16 + half * 8) * N + tn * 16 + (lane & 15);
#pragma unroll
    for (int r = 0; r < 8; ++r) cptr[(long)r * N] = acc[r];
}

// ---------------------------------------------------------------------------
// agg[n, :] = b[:]  (broadcast init so the scatter can accumulate on top)
// ---------------------------------------------------------------------------
__global__ void init_bias(float* __restrict__ out, const float* __restrict__ b,
                          int nmask, long total) {
    long i = (long)blockIdx.x * blockDim.x + threadIdx.x;
    long stride = (long)gridDim.x * blockDim.x;
    for (; i < total; i += stride) out[i] = b[i & nmask];
}

// ---------------------------------------------------------------------------
// COO scatter: one wave per edge; lane strides the feature dim.
// agg[row[e], f] += ew[e] * support[col[e], f]   (fp32 atomics, L2-resident)
// ---------------------------------------------------------------------------
__global__ void scatter_edges(const int* __restrict__ row, const int* __restrict__ col,
                              const float* __restrict__ ew,
                              const float* __restrict__ support,
                              float* __restrict__ agg, int E, int N) {
    int wid  = (int)(((long)blockIdx.x * blockDim.x + threadIdx.x) >> 5);
    int lane = threadIdx.x & 31;
    if (wid >= E) return;
    int r = row[wid];
    int c = col[wid];
    float w = ew[wid];
    const float* s = support + (long)c * N;
    float* o = agg + (long)r * N;
    for (int f = lane; f < N; f += 32) atomicAdd(o + f, w * s[f]);
}

__global__ void zero_f32(float* __restrict__ p, int n) {
    int i = blockIdx.x * blockDim.x + threadIdx.x;
    if (i < n) p[i] = 0.0f;
}

// ---------------------------------------------------------------------------
// BN stats over relu(x): per-column sum & sumsq. blockDim.x == Nfeat (256).
// ---------------------------------------------------------------------------
__global__ void bn_stats(const float* __restrict__ x, float* __restrict__ sum,
                         float* __restrict__ sumsq, int rowsPerBlock, int Nfeat,
                         int totalRows) {
    int c  = threadIdx.x;
    int r0 = blockIdx.x * rowsPerBlock;
    int r1 = r0 + rowsPerBlock;
    if (r1 > totalRows) r1 = totalRows;
    float s = 0.0f, s2 = 0.0f;
    for (int r = r0; r < r1; ++r) {
        float v = x[(long)r * Nfeat + c];
        v = v > 0.0f ? v : 0.0f;
        s += v;
        s2 += v * v;
    }
    atomicAdd(&sum[c], s);
    atomicAdd(&sumsq[c], s2);
}

// ---------------------------------------------------------------------------
// y = bf16( (relu(x) - mean) * rsqrt(var + eps) )  -> feeds the next GEMM
// ---------------------------------------------------------------------------
__global__ void bn_norm_bf16(const float* __restrict__ x, const float* __restrict__ sum,
                             const float* __restrict__ sumsq, __bf16* __restrict__ y,
                             int nmask, long total, float invN) {
    long i = (long)blockIdx.x * blockDim.x + threadIdx.x;
    long stride = (long)gridDim.x * blockDim.x;
    for (; i < total; i += stride) {
        int c = (int)(i & nmask);
        float mean = sum[c] * invN;
        float var  = sumsq[c] * invN - mean * mean;
        float v = x[i];
        v = v > 0.0f ? v : 0.0f;
        y[i] = f2bf((v - mean) * rsqrtf(var + BN_EPS));
    }
}

__global__ void relu_inplace(float* __restrict__ p, long n) {
    long i = (long)blockIdx.x * blockDim.x + threadIdx.x;
    long stride = (long)gridDim.x * blockDim.x;
    for (; i < n; i += stride) {
        float v = p[i];
        p[i] = v > 0.0f ? v : 0.0f;
    }
}

// ---------------------------------------------------------------------------
extern "C" void kernel_launch(void* const* d_in, const int* in_sizes, int n_in,
                              void* d_out, int out_size, void* d_ws, size_t ws_size,
                              hipStream_t stream) {
    const float* x   = (const float*)d_in[0];
    const int*   eix = (const int*)d_in[1];
    const float* ew  = (const float*)d_in[2];
    const float* W1  = (const float*)d_in[3];
    const float* b1  = (const float*)d_in[4];
    const float* W2  = (const float*)d_in[5];
    const float* b2  = (const float*)d_in[6];
    const float* W3  = (const float*)d_in[7];
    const float* b3  = (const float*)d_in[8];
    float* out = (float*)d_out;

    const int E = in_sizes[2];
    const int* row = eix;
    const int* col = eix + E;

    const int NF = 512, NH = 256, NC = 64;

    // workspace carve-up (256B aligned)
    char* ws = (char*)d_ws;
    size_t off = 0;
    auto carve = [&](size_t bytes) -> void* {
        void* p = ws + off;
        off = (off + bytes + 255) & ~(size_t)255;
        return p;
    };
    __bf16* Abf     = (__bf16*)carve((size_t)NNODES * NF * sizeof(__bf16)); // bf16 activations
    __bf16* Wp1     = (__bf16*)carve((size_t)NF * NH * sizeof(__bf16));
    __bf16* Wp2     = (__bf16*)carve((size_t)NH * NH * sizeof(__bf16));
    __bf16* Wp3     = (__bf16*)carve((size_t)NH * NC * sizeof(__bf16));
    float*  support = (float*)carve((size_t)NNODES * NH * sizeof(float));
    float*  agg     = (float*)carve((size_t)NNODES * NH * sizeof(float));
    float*  stat_s  = (float*)carve(256 * sizeof(float));
    float*  stat_q  = (float*)carve(256 * sizeof(float));

    const int TPB = 256;
    const int edgeBlocks = (E + 7) / 8;                 // 8 waves/block, 1 edge/wave
    const int bnRows = 250;
    const int bnBlocks = (NNODES + bnRows - 1) / bnRows;
    const float invN = 1.0f / (float)NNODES;

    // activations -> bf16
    cvt_f32_bf16<<<4096, TPB, 0, stream>>>(x, Abf, (long)NNODES * NF);
    // pack all weights into WMMA B-fragment order
    pack_w_bf16<<<(NF * NH + TPB - 1) / TPB, TPB, 0, stream>>>(W1, Wp1, NF, NH);
    pack_w_bf16<<<(NH * NH + TPB - 1) / TPB, TPB, 0, stream>>>(W2, Wp2, NH, NH);
    pack_w_bf16<<<(NH * NC + TPB - 1) / TPB, TPB, 0, stream>>>(W3, Wp3, NH, NC);

    // ---- layer 1: support = X @ W1 ; agg = b1 + scatter ; h = BN(relu(agg)) ----
    {
        int tiles = (NNODES / 16) * (NH / 16);
        wmma_gemm_bf16<<<(tiles + 7) / 8, TPB, 0, stream>>>(Abf, Wp1, support,
                                                            NNODES, NH, NF);
        init_bias<<<8192, TPB, 0, stream>>>(agg, b1, NH - 1, (long)NNODES * NH);
        scatter_edges<<<edgeBlocks, TPB, 0, stream>>>(row, col, ew, support, agg, E, NH);
        zero_f32<<<2, TPB, 0, stream>>>(stat_s, NH);
        zero_f32<<<2, TPB, 0, stream>>>(stat_q, NH);
        bn_stats<<<bnBlocks, NH, 0, stream>>>(agg, stat_s, stat_q, bnRows, NH, NNODES);
        bn_norm_bf16<<<8192, TPB, 0, stream>>>(agg, stat_s, stat_q, Abf,
                                               NH - 1, (long)NNODES * NH, invN);
    }
    // ---- layer 2 ----
    {
        int tiles = (NNODES / 16) * (NH / 16);
        wmma_gemm_bf16<<<(tiles + 7) / 8, TPB, 0, stream>>>(Abf, Wp2, support,
                                                            NNODES, NH, NH);
        init_bias<<<8192, TPB, 0, stream>>>(agg, b2, NH - 1, (long)NNODES * NH);
        scatter_edges<<<edgeBlocks, TPB, 0, stream>>>(row, col, ew, support, agg, E, NH);
        zero_f32<<<2, TPB, 0, stream>>>(stat_s, NH);
        zero_f32<<<2, TPB, 0, stream>>>(stat_q, NH);
        bn_stats<<<bnBlocks, NH, 0, stream>>>(agg, stat_s, stat_q, bnRows, NH, NNODES);
        bn_norm_bf16<<<8192, TPB, 0, stream>>>(agg, stat_s, stat_q, Abf,
                                               NH - 1, (long)NNODES * NH, invN);
    }
    // ---- layer 3: aggregate straight into d_out, final relu ----
    {
        int tiles = (NNODES / 16) * (NC / 16);
        wmma_gemm_bf16<<<(tiles + 7) / 8, TPB, 0, stream>>>(Abf, Wp3, support,
                                                            NNODES, NC, NH);
        init_bias<<<8192, TPB, 0, stream>>>(out, b3, NC - 1, (long)NNODES * NC);
        scatter_edges<<<edgeBlocks, TPB, 0, stream>>>(row, col, ew, support, out, E, NC);
        relu_inplace<<<8192, TPB, 0, stream>>>(out, (long)NNODES * NC);
    }
}
